// ChainOfThought_30958124270230
// MI455X (gfx1250) — matrix operand order repaired and checked by reading, not compiled, not run on hardware
//
#include <hip/hip_runtime.h>

// Problem constants (match reference)
#define B_    2048
#define DIN_  1024
#define E_    1024
#define F_    16384
#define NH_   8
#define DH_   128
#define STEPS_ 3

typedef unsigned short u16;
typedef __attribute__((ext_vector_type(4)))  u16    v4u16;
typedef __attribute__((ext_vector_type(8)))  u16    v8u16;
typedef __attribute__((ext_vector_type(16))) u16    v16u16;
typedef __attribute__((ext_vector_type(16))) __bf16 v16bf;
typedef __attribute__((ext_vector_type(8)))  float  v8f;
typedef __attribute__((ext_vector_type(4)))  unsigned int v4u;
typedef __attribute__((ext_vector_type(4)))  int    v4i;
typedef __attribute__((ext_vector_type(8)))  int    v8i;

__device__ __forceinline__ u16 f2bf(float f) {
  unsigned int u = __float_as_uint(f);
  u += 0x7FFFu + ((u >> 16) & 1u);   // round-to-nearest-even bf16
  return (u16)(u >> 16);
}
__device__ __forceinline__ float bf2f(u16 b) {
  return __uint_as_float(((unsigned int)b) << 16);
}
__device__ __forceinline__ v4u16 to_bf4(const float4& v) {
  return (v4u16){ f2bf(v.x), f2bf(v.y), f2bf(v.z), f2bf(v.w) };
}
__device__ __forceinline__ void  storeC(float* p, float v) { *p = v; }
__device__ __forceinline__ void  storeC(u16* p, float v)   { *p = f2bf(v); }
__device__ __forceinline__ float loadC(const float* p)     { return *p; }
__device__ __forceinline__ float loadC(const u16* p)       { return bf2f(*p); }

// ---------------------------------------------------------------------------
// TDM: async-load a 2D bf16 tile [128 rows x 64 cols] (row stride = lda elts)
// from global into LDS at lds_off, padding each 128B row with 16B so the LDS
// image has 72-u16 row pitch. D# per ISA 8.3-8.6: data_size=2B,
// pad_interval=32 DWORDs (128B), pad_amount=4 DWORDs (16B),
// tile_dim0=64, tile_dim1=128. Wave-level op; issue from one wave only.
// ---------------------------------------------------------------------------
__device__ __forceinline__ void tdm_load_tile(unsigned lds_off, const u16* gaddr,
                                              unsigned lda_elts) {
  unsigned long long ga = (unsigned long long)gaddr;
  v4u g0 = { 1u,                                            // count=1 (user D#)
             lds_off,                                       // lds_addr
             (unsigned)(ga & 0xFFFFFFFFu),                  // global_addr[31:0]
             (unsigned)((ga >> 32) & 0x01FFFFFFu) | 0x80000000u }; // addr[56:32] | type=2
  const unsigned d0 = (1u << 16)    // data_size = 1 -> 2 bytes
                    | (1u << 20)    // pad_enable
                    | (4u << 22)    // pad_interval: 32 DWORDs (128B)
                    | (3u << 25);   // pad_amount: 4 DWORDs (16B)
  const unsigned td0 = 0x40000000u; // huge tensor dims: never OOB
  const unsigned td1 = 0x40000000u;
  v8i g1 = { (int)d0,
             (int)((td0 & 0xFFFFu) << 16),                      // tensor_dim0[15:0]
             (int)((td0 >> 16) | ((td1 & 0xFFFFu) << 16)),      // dim0 hi | dim1 lo
             (int)((td1 >> 16) | (64u << 16)),                  // dim1 hi | tile_dim0=64
             (int)(128u),                                       // tile_dim1=128, tile_dim2=0
             (int)(lda_elts),                                   // tensor_dim0_stride lo
             0, 0 };                                            // stride hi / dim1 stride
  v4i z4 = {0, 0, 0, 0};
#if defined(__clang_major__) && (__clang_major__ >= 23)
  v8i z8 = {0, 0, 0, 0, 0, 0, 0, 0};
  __builtin_amdgcn_tensor_load_to_lds(g0, g1, z4, z4, z8, 0);
#else
  __builtin_amdgcn_tensor_load_to_lds(g0, g1, z4, z4, 0);
#endif
}

// ---------------------------------------------------------------------------
// GEMM:  C[M,N] = act( A[M,K]bf16 (row stride lda) @ W[N,K]bf16^T + bias )(*mul)
// v_wmma_f32_16x16x32_bf16; block tile 128x128, 8 waves (4M x 2N), wave tile
// 32x64. K-tile 64, 3-deep LDS ring fed by TDM; one barrier + one
// s_wait_tensorcnt per 64-K step; DMA keeps 2 tiles (128 K) in flight.
// M%128==0, N%128==0, K%64==0 (all shapes comply).
// ---------------------------------------------------------------------------
template <typename TC, bool RELU, bool HAS_MUL>
__global__ __launch_bounds__(256) void gemm_wmma(
    const u16* __restrict__ A, int lda,
    const u16* __restrict__ W,                 // [N,K] bf16, row stride K
    const float* __restrict__ bias,            // [N]
    TC* __restrict__ C, int ldc,
    const TC* __restrict__ mul,                // elementwise factor (if HAS_MUL)
    int K)
{
  constexpr int LDSS = 72;                     // u16 per LDS row (64 data + 8 pad)
  __shared__ __align__(16) u16 As[3][128 * LDSS];
  __shared__ __align__(16) u16 Bs[3][128 * LDSS];

  const int t     = threadIdx.x;
  const int lane  = t & 31;
  const int wave  = t >> 5;        // 0..7
  const int waveM = wave & 3;      // 4 waves along M
  const int waveN = wave >> 2;     // 2 waves along N
  const int h     = lane >> 4;     // lane half
  const int r     = lane & 15;
  const bool w0   = ((__builtin_amdgcn_readfirstlane(t) >> 5) == 0); // scalar guard

  const int bM = blockIdx.y * 128;
  const int bN = blockIdx.x * 128;
  const u16* Abase = A + (size_t)bM * lda;
  const u16* Wbase = W + (size_t)bN * K;

  const v8f vzero = {0.f, 0.f, 0.f, 0.f, 0.f, 0.f, 0.f, 0.f};
  v8f acc[2][4];
#pragma unroll
  for (int mi = 0; mi < 2; ++mi)
#pragma unroll
    for (int ni = 0; ni < 4; ++ni) acc[mi][ni] = vzero;

  const int nk = K >> 6;                       // 64-K tiles (nk >= 16 here)
  if (w0) {
    tdm_load_tile((unsigned)(unsigned long long)&As[0][0], Abase, (unsigned)lda);
    tdm_load_tile((unsigned)(unsigned long long)&Bs[0][0], Wbase, (unsigned)K);
    tdm_load_tile((unsigned)(unsigned long long)&As[1][0], Abase + 64, (unsigned)lda);
    tdm_load_tile((unsigned)(unsigned long long)&Bs[1][0], Wbase + 64, (unsigned)K);
  }

  for (int ks = 0; ks < nk; ++ks) {
    const int buf = ks % 3;
    // retire tile ks (tile ks+1 may stay in flight); no-op for waves 1..7
    if (ks + 1 < nk) __builtin_amdgcn_s_wait_tensorcnt(2);
    else             __builtin_amdgcn_s_wait_tensorcnt(0);
    __syncthreads();   // tile ks visible to all; ring slot (ks+2)%3 free of readers
    if (w0 && ks + 2 < nk) {
      const int nbuf = (ks + 2) % 3;
      tdm_load_tile((unsigned)(unsigned long long)&As[nbuf][0],
                    Abase + (ks + 2) * 64, (unsigned)lda);
      tdm_load_tile((unsigned)(unsigned long long)&Bs[nbuf][0],
                    Wbase + (ks + 2) * 64, (unsigned)K);
    }
    // prefetch past the TDM window into GL2 (global_prefetch_b8)
    if (ks + 3 < nk) {
      __builtin_prefetch(Abase + (size_t)(t >> 1) * lda + (ks + 3) * 64 + (t & 1) * 32, 0, 1);
      __builtin_prefetch(Wbase + (size_t)(t >> 1) * K + (ks + 3) * 64 + (t & 1) * 32, 0, 1);
    }

#pragma unroll
    for (int kk = 0; kk < 64; kk += 32) {
      // A fragment (16x32 bf16): e[0..7]=K(8h..+7), e[8..15]=K(16+8h..+7)
      v16bf afrag[2];
#pragma unroll
      for (int mi = 0; mi < 2; ++mi) {
        const u16* rowp = &As[buf][(waveM * 32 + mi * 16 + r) * LDSS + kk];
        v8u16 a0 = *(const v8u16*)(rowp + h * 8);
        v8u16 a1 = *(const v8u16*)(rowp + 16 + h * 8);
        v16u16 au = __builtin_shufflevector(a0, a1, 0,1,2,3,4,5,6,7,8,9,10,11,12,13,14,15);
        afrag[mi] = __builtin_bit_cast(v16bf, au);
      }
      // B fragment (32x16 bf16): e[0..7]=K(8h..+7), e[8..15]=K(16+8h..+7)
      v16bf bfrag[4];
#pragma unroll
      for (int ni = 0; ni < 4; ++ni) {
        const u16* rowp = &Bs[buf][(waveN * 64 + ni * 16 + r) * LDSS + kk];
        v8u16 b0 = *(const v8u16*)(rowp + 8 * h);
        v8u16 b1 = *(const v8u16*)(rowp + 16 + 8 * h);
        v16u16 bu = __builtin_shufflevector(b0, b1, 0,1,2,3,4,5,6,7,8,9,10,11,12,13,14,15);
        bfrag[ni] = __builtin_bit_cast(v16bf, bu);
      }
#pragma unroll
      for (int mi = 0; mi < 2; ++mi)
#pragma unroll
        for (int ni = 0; ni < 4; ++ni)
          acc[mi][ni] = __builtin_amdgcn_wmma_f32_16x16x32_bf16(
              false, afrag[mi], false, bfrag[ni], (short)0, acc[mi][ni], false, false);
    }
  }

  // Epilogue (straight-line; D layout: VGPR v -> row M=v+8h, col N=r)
#pragma unroll
  for (int mi = 0; mi < 2; ++mi) {
#pragma unroll
    for (int ni = 0; ni < 4; ++ni) {
      const int col = bN + waveN * 64 + ni * 16 + r;
      const float bv = bias[col];
#pragma unroll
      for (int v = 0; v < 8; ++v) {
        const int row = bM + waveM * 32 + mi * 16 + h * 8 + v;
        float val = acc[mi][ni][v] + bv;
        if (RELU) val = fmaxf(val, 0.0f);
        const size_t idx = (size_t)row * ldc + col;
        if (HAS_MUL) val *= loadC(mul + idx);
        storeC(C + idx, val);
      }
    }
  }
}

// ---------------------------------------------------------------------------
// f32 -> bf16 bulk convert (n multiple of 8)
// ---------------------------------------------------------------------------
__global__ __launch_bounds__(256) void cvt_bf16(const float* __restrict__ in,
                                                u16* __restrict__ out,
                                                long long n) {
  long long i = ((long long)blockIdx.x * 256 + threadIdx.x) * 8;
  if (i + 8 <= n) {
    float4 a = *(const float4*)(in + i);
    float4 b = *(const float4*)(in + i + 4);
    v4u16 pa = to_bf4(a), pb = to_bf4(b);
    v8u16 p = __builtin_shufflevector(pa, pb, 0,1,2,3,4,5,6,7);
    *(v8u16*)(out + i) = p;
  }
}

// ---------------------------------------------------------------------------
// LayerNorm over rows of length E_=1024, fused residual add; writes f32 and
// bf16 copies, plus optional second strided outputs (ctx slot).
//   y = (HASRES? res:0) + LN(X)*g + b
// ---------------------------------------------------------------------------
template <bool HASRES, bool HAS2>
__global__ __launch_bounds__(256) void ln_fuse(
    const float* __restrict__ X, const float* __restrict__ res,
    const float* __restrict__ g, const float* __restrict__ bta,
    float* __restrict__ Yf, u16* __restrict__ Ybf,
    float* __restrict__ Y2f, u16* __restrict__ Y2bf, int ld2)
{
  const int row = blockIdx.x;
  const int t = threadIdx.x;
  const float* x = X + (size_t)row * E_;
  float v[4];
  float s = 0.f;
#pragma unroll
  for (int i = 0; i < 4; ++i) { v[i] = x[t + i * 256]; s += v[i]; }
  __shared__ float red[8];
#pragma unroll
  for (int off = 16; off > 0; off >>= 1) s += __shfl_xor(s, off, 32);
  const int wv = t >> 5, lane = t & 31;
  if (lane == 0) red[wv] = s;
  __syncthreads();
  float tot = 0.f;
#pragma unroll
  for (int i = 0; i < 8; ++i) tot += red[i];
  const float mean = tot * (1.0f / E_);
  float vs = 0.f;
#pragma unroll
  for (int i = 0; i < 4; ++i) { float d = v[i] - mean; vs += d * d; }
#pragma unroll
  for (int off = 16; off > 0; off >>= 1) vs += __shfl_xor(vs, off, 32);
  __syncthreads();
  if (lane == 0) red[wv] = vs;
  __syncthreads();
  float vtot = 0.f;
#pragma unroll
  for (int i = 0; i < 8; ++i) vtot += red[i];
  const float rstd = rsqrtf(vtot * (1.0f / E_) + 1e-5f);
#pragma unroll
  for (int i = 0; i < 4; ++i) {
    const int c = t + i * 256;
    float y = (v[i] - mean) * rstd * g[c] + bta[c];
    if (HASRES) y += res[(size_t)row * E_ + c];
    Yf[(size_t)row * E_ + c] = y;
    Ybf[(size_t)row * E_ + c] = f2bf(y);
    if (HAS2) {
      Y2f[(size_t)row * ld2 + c] = y;
      Y2bf[(size_t)row * ld2 + c] = f2bf(y);
    }
  }
}

// ---------------------------------------------------------------------------
// Cross-attention with tiny context (L<=3). One wave per (b,h); lane covers
// 4 of the 128 head dims. K/V stored [B,4,E] f32. Output written bf16.
// ---------------------------------------------------------------------------
__global__ __launch_bounds__(256) void attn_small(
    const float* __restrict__ Q,    // [B,E]
    const float* __restrict__ Kb,   // [B,4,E]
    const float* __restrict__ Vb,   // [B,4,E]
    u16* __restrict__ O,            // [B,E] bf16
    int L)
{
  const int wave = threadIdx.x >> 5, lane = threadIdx.x & 31;
  const int gw = blockIdx.x * 8 + wave;
  const int b = gw >> 3;
  const int hh = gw & 7;
  const float* qp = Q + (size_t)b * E_ + hh * DH_;
  float qv[4];
#pragma unroll
  for (int i = 0; i < 4; ++i) qv[i] = qp[lane + i * 32];

  float scores[3];
  for (int l = 0; l < L; ++l) {
    const float* kp = Kb + (size_t)b * 4 * E_ + (size_t)l * E_ + hh * DH_;
    float p = 0.f;
#pragma unroll
    for (int i = 0; i < 4; ++i) p += qv[i] * kp[lane + i * 32];
#pragma unroll
    for (int off = 16; off > 0; off >>= 1) p += __shfl_xor(p, off, 32);
    scores[l] = p * 0.08838834764831845f;   // 1/sqrt(128)
  }
  float mx = scores[0];
  for (int l = 1; l < L; ++l) mx = fmaxf(mx, scores[l]);
  float e[3], se = 0.f;
  for (int l = 0; l < L; ++l) { e[l] = __expf(scores[l] - mx); se += e[l]; }
  const float inv = 1.f / se;
  float o[4] = {0.f, 0.f, 0.f, 0.f};
  for (int l = 0; l < L; ++l) {
    const float* vp = Vb + (size_t)b * 4 * E_ + (size_t)l * E_ + hh * DH_;
    const float w = e[l] * inv;
#pragma unroll
    for (int i = 0; i < 4; ++i) o[i] += w * vp[lane + i * 32];
  }
  u16* op = O + (size_t)b * E_ + hh * DH_;
#pragma unroll
  for (int i = 0; i < 4; ++i) op[lane + i * 32] = f2bf(o[i]);
}

// ---------------------------------------------------------------------------
template <typename TC, bool RELU, bool HAS_MUL>
static void launch_gemm(const u16* A, int lda, const u16* W, const float* bias,
                        TC* C, int ldc, const TC* mul, int M, int N, int K,
                        hipStream_t stream) {
  dim3 grid(N / 128, M / 128), blk(256);
  gemm_wmma<TC, RELU, HAS_MUL><<<grid, blk, 0, stream>>>(A, lda, W, bias, C, ldc, mul, K);
}

static void cvt(const float* in, u16* out, long long n, hipStream_t stream) {
  dim3 blk(256);
  cvt_bf16<<<(unsigned)(n / (256 * 8)), blk, 0, stream>>>(in, out, n);
}

extern "C" void kernel_launch(void* const* d_in, const int* in_sizes, int n_in,
                              void* d_out, int out_size, void* d_ws, size_t ws_size,
                              hipStream_t stream) {
  (void)in_sizes; (void)n_in; (void)out_size; (void)ws_size;
  const float* image = (const float*)d_in[0];
  const float* text  = (const float*)d_in[1];
  const float* fc1_w = (const float*)d_in[2];
  const float* fc1_b = (const float*)d_in[3];
  const float* fc2_w = (const float*)d_in[4];
  const float* fc2_b = (const float*)d_in[5];
  const float* fc3_w = (const float*)d_in[6];
  const float* fc3_b = (const float*)d_in[7];
  const float* ln_g  = (const float*)d_in[8];
  const float* ln_b  = (const float*)d_in[9];
  const float* Wqkv  = (const float*)d_in[10];
  const float* bqkv  = (const float*)d_in[11];
  const float* Wo    = (const float*)d_in[12];
  const float* bo    = (const float*)d_in[13];
  const float* W1    = (const float*)d_in[14];
  const float* b1    = (const float*)d_in[15];
  const float* W2    = (const float*)d_in[16];
  const float* b2    = (const float*)d_in[17];
  const float* n1g   = (const float*)d_in[18];
  const float* n1b   = (const float*)d_in[19];
  const float* n2g   = (const float*)d_in[20];
  const float* n2b   = (const float*)d_in[21];
  const float* Wout  = (const float*)d_in[22];
  const float* bout  = (const float*)d_in[23];

  float* out_final = (float*)d_out;                       // [B, DIN]
  float* ctx = out_final + (size_t)B_ * DIN_;             // [B, 4, E] (2nd output)

  // ---- workspace: bf16 region then f32 region ----
  u16* p16 = (u16*)d_ws;
  u16* fc1wb = p16;  p16 += (size_t)F_ * DIN_;
  u16* fc2wb = p16;  p16 += (size_t)F_ * DIN_;
  u16* fc3wb = p16;  p16 += (size_t)E_ * F_;
  u16* Wqkvb = p16;  p16 += (size_t)STEPS_ * 3 * E_ * E_;
  u16* Wob   = p16;  p16 += (size_t)STEPS_ * E_ * E_;
  u16* W1b   = p16;  p16 += (size_t)STEPS_ * E_ * E_;
  u16* W2b   = p16;  p16 += (size_t)STEPS_ * E_ * E_;
  u16* Woutb = p16;  p16 += (size_t)DIN_ * E_;
  u16* imgb  = p16;  p16 += (size_t)B_ * DIN_;
  u16* txtb  = p16;  p16 += (size_t)B_ * DIN_;
  u16* t1    = p16;  p16 += (size_t)B_ * F_;              // fused [B,F] bf16
  u16* xbf   = p16;  p16 += (size_t)B_ * E_;
  u16* ctxbf = p16;  p16 += (size_t)4 * B_ * E_;
  u16* h1bf  = p16;  p16 += (size_t)B_ * E_;
  float* pf = (float*)p16;
  float* xf = pf;    pf += (size_t)B_ * E_;
  float* qf = pf;    pf += (size_t)B_ * E_;
  float* kbuf = pf;  pf += (size_t)4 * B_ * E_;
  float* vbuf = pf;  pf += (size_t)4 * B_ * E_;
  float* h2 = pf;    pf += (size_t)B_ * E_;

  dim3 blk(256);

  // ---- one-time bf16 conversions (copies stay L2-resident afterwards) ----
  cvt(image, imgb, (long long)B_ * DIN_, stream);
  cvt(text,  txtb, (long long)B_ * DIN_, stream);
  cvt(fc1_w, fc1wb, (long long)F_ * DIN_, stream);
  cvt(fc2_w, fc2wb, (long long)F_ * DIN_, stream);
  cvt(fc3_w, fc3wb, (long long)E_ * F_, stream);
  cvt(Wqkv,  Wqkvb, (long long)STEPS_ * 3 * E_ * E_, stream);
  cvt(Wo,    Wob,   (long long)STEPS_ * E_ * E_, stream);
  cvt(W1,    W1b,   (long long)STEPS_ * E_ * E_, stream);
  cvt(W2,    W2b,   (long long)STEPS_ * E_ * E_, stream);
  cvt(Wout,  Woutb, (long long)DIN_ * E_, stream);

  // ---- fusion MLP ----
  launch_gemm<u16, false, false>(imgb, DIN_, fc1wb, fc1_b, t1, F_, (u16*)nullptr,
                                 B_, F_, DIN_, stream);
  launch_gemm<u16, false, true>(txtb, DIN_, fc2wb, fc2_b, t1, F_, t1,
                                B_, F_, DIN_, stream);
  launch_gemm<float, false, false>(t1, F_, fc3wb, fc3_b, h2, E_, (float*)nullptr,
                                   B_, E_, F_, stream);
  // fused = LN(h2) -> xf/xbf and ctx slot 0 (f32 output + bf16 copy)
  ln_fuse<false, true><<<B_, blk, 0, stream>>>(h2, nullptr, ln_g, ln_b,
                                               xf, xbf, ctx, ctxbf, 4 * E_);

  // ---- transformer steps ----
  for (int i = 0; i < STEPS_; ++i) {
    const int L = i + 1;
    const u16* Wqb = Wqkvb + (size_t)i * 3 * E_ * E_;
    const float* bq = bqkv + (size_t)i * 3 * E_;
    launch_gemm<float, false, false>(xbf, E_, Wqb, bq, qf, E_, (float*)nullptr,
                                     B_, E_, E_, stream);
    for (int l = 0; l < L; ++l) {
      launch_gemm<float, false, false>(ctxbf + (size_t)l * E_, 4 * E_,
          Wqb + (size_t)E_ * E_, bq + E_, kbuf + (size_t)l * E_, 4 * E_,
          (float*)nullptr, B_, E_, E_, stream);
      launch_gemm<float, false, false>(ctxbf + (size_t)l * E_, 4 * E_,
          Wqb + (size_t)2 * E_ * E_, bq + 2 * E_, vbuf + (size_t)l * E_, 4 * E_,
          (float*)nullptr, B_, E_, E_, stream);
    }
    attn_small<<<B_ * NH_ / 8, blk, 0, stream>>>(qf, kbuf, vbuf, h1bf, L);
    launch_gemm<float, false, false>(h1bf, E_, Wob + (size_t)i * E_ * E_,
        bo + (size_t)i * E_, h2, E_, (float*)nullptr, B_, E_, E_, stream);
    // x = x + LN(attn)
    ln_fuse<true, false><<<B_, blk, 0, stream>>>(h2, xf,
        n1g + (size_t)i * E_, n1b + (size_t)i * E_, xf, xbf, nullptr, nullptr, 0);
    // ffn: h1 = relu(x@W1^T+b1) (bf16), h2 = h1@W2^T+b2 (f32)
    launch_gemm<u16, true, false>(xbf, E_, W1b + (size_t)i * E_ * E_,
        b1 + (size_t)i * E_, h1bf, E_, (u16*)nullptr, B_, E_, E_, stream);
    launch_gemm<float, false, false>(h1bf, E_, W2b + (size_t)i * E_ * E_,
        b2 + (size_t)i * E_, h2, E_, (float*)nullptr, B_, E_, E_, stream);
    // x = x + LN(ffn); also append into ctx slot i+1 (f32 + bf16)
    ln_fuse<true, true><<<B_, blk, 0, stream>>>(h2, xf,
        n2g + (size_t)i * E_, n2b + (size_t)i * E_, xf, xbf,
        ctx + (size_t)(i + 1) * E_, ctxbf + (size_t)(i + 1) * E_, 4 * E_);
  }

  // ---- final projection ----
  launch_gemm<float, false, false>(xbf, E_, Woutb, bout, out_final, DIN_,
                                   (float*)nullptr, B_, DIN_, E_, stream);
}
